// MultiHeadAttention_9354438771380
// MI455X (gfx1250) — matrix operand order repaired and checked by reading, not compiled, not run on hardware
//
#include <hip/hip_runtime.h>
#include <hip/hip_bf16.h>
#include <math.h>

// ---------------------------------------------------------------------------
// MHA for B=4, T=2048, D=1024, H=16, DK=64 on gfx1250 (CDNA5, wave32, WMMA).
// f32->bf16 convert, weight transposes, pipelined WMMA QKV GEMM,
// LDS-staged flash-attention (async global->LDS copies, online softmax, WMMA),
// pipelined WMMA projection GEMM.
// ---------------------------------------------------------------------------

typedef __bf16 bf16_t;
typedef __attribute__((ext_vector_type(16))) __bf16 v16bf;
typedef __attribute__((ext_vector_type(8)))  __bf16 v8bf;
typedef __attribute__((ext_vector_type(4)))  __bf16 v4bf;
typedef __attribute__((ext_vector_type(8)))  float  v8f;
typedef int v4i __attribute__((vector_size(16)));   // matches async builtin proto

#define Bq 4
#define Tq 2048
#define Dq 1024
#define Hq 16
#define DKq 64

// K-tile rows padded to 72 elems (144B = 36 banks), V-tile rows to 40 elems
// (80B = 20 banks): both give 16 distinct banks across the 16-lane frag read.
#define KT_STRIDE 72
#define VT_STRIDE 40

#if defined(__has_builtin)
#if __has_builtin(__builtin_amdgcn_global_load_async_to_lds_b128) && \
    __has_builtin(__builtin_amdgcn_s_wait_asynccnt)
#define ASYNC_COPY_OK 1
#endif
#endif

#define GAS __attribute__((address_space(1)))
#define LAS __attribute__((address_space(3)))

__device__ __forceinline__ v8f wmma_bf16(v16bf a, v16bf b, v8f c) {
    return __builtin_amdgcn_wmma_f32_16x16x32_bf16(
        /*neg_a=*/false, a, /*neg_b=*/false, b,
        /*c_mod=*/(short)0, c, /*reuse_a=*/false, /*reuse_b=*/false);
}

// A-matrix 16x32 bf16 fragment (ISA 7.12.2): lane L (m=L&15, hh=L>>4) holds
// K = {hh*8 .. hh*8+7} in elems 0..7 and K = {16+hh*8 ..} in elems 8..15.
__device__ __forceinline__ v16bf load_A_frag(const bf16_t* base, int row_stride, int lane) {
    const int m = lane & 15, hh = lane >> 4;
    const bf16_t* p = base + m * row_stride + hh * 8;
    v8bf lo = *(const v8bf*)(p);
    v8bf hi = *(const v8bf*)(p + 16);
    v16bf r;
    #pragma unroll
    for (int i = 0; i < 8; ++i) { r[i] = lo[i]; r[i + 8] = hi[i]; }
    return r;
}

// B-matrix 32x16 bf16 fragment: lane L (n=L&15, hh=L>>4) holds column n,
// K = {hh*8 .. hh*8+7} in elems 0..7 and K = {16+hh*8 ..} in elems 8..15.
__device__ __forceinline__ v16bf load_B_frag(const bf16_t* base, int row_stride, int lane) {
    const int n = lane & 15, hh = lane >> 4;
    const bf16_t* p = base + n * row_stride + hh * 8;
    v8bf lo = *(const v8bf*)(p);
    v8bf hi = *(const v8bf*)(p + 16);
    v16bf r;
    #pragma unroll
    for (int i = 0; i < 8; ++i) { r[i] = lo[i]; r[i + 8] = hi[i]; }
    return r;
}

// ---------------------------------------------------------------------------
// Elementwise f32 -> bf16 (x input), vectorized x4.
// ---------------------------------------------------------------------------
__global__ __launch_bounds__(256) void cvt_f32_bf16(const float* __restrict__ in,
                                                    bf16_t* __restrict__ out, int n) {
    int idx = (blockIdx.x * 256 + threadIdx.x) * 4;
    if (idx + 3 < n) {
        float4 v = *(const float4*)(in + idx);
        v4bf o;
        o[0] = (bf16_t)v.x; o[1] = (bf16_t)v.y; o[2] = (bf16_t)v.z; o[3] = (bf16_t)v.w;
        *(v4bf*)(out + idx) = o;
    }
}

// ---------------------------------------------------------------------------
// Transpose f32 [R][C] -> bf16 [C][R] via LDS tile (coalesced both sides).
// ---------------------------------------------------------------------------
__global__ __launch_bounds__(256) void transpose_to_bf16(const float* __restrict__ in,
                                                         bf16_t* __restrict__ out,
                                                         int R, int C) {
    __shared__ float tile[32][33];
    const int c0 = blockIdx.x * 32, r0 = blockIdx.y * 32;
    const int tx = threadIdx.x, ty = threadIdx.y;   // 32 x 8
    #pragma unroll
    for (int i = 0; i < 32; i += 8) {
        int r = r0 + ty + i, c = c0 + tx;
        tile[ty + i][tx] = (r < R && c < C) ? in[(size_t)r * C + c] : 0.f;
    }
    __syncthreads();
    #pragma unroll
    for (int i = 0; i < 32; i += 8) {
        int c = c0 + ty + i, r = r0 + tx;
        if (c < C && r < R) out[(size_t)c * R + r] = (bf16_t)tile[tx][ty + i];
    }
}

// ---------------------------------------------------------------------------
// Pipelined GEMM core: wave tile 32M x 64N (8 f32 accs), K-step 32, register
// double-buffering so next step's loads overlap current step's 8 WMMAs.
// ---------------------------------------------------------------------------
__device__ __forceinline__ void gemm_core_32x64(const bf16_t* __restrict__ A,
                                                const bf16_t* __restrict__ Wt,
                                                int m0, int n0, int lane, v8f acc[8]) {
    const bf16_t* Ar0 = A + (size_t)m0 * Dq;
    const bf16_t* Ar1 = A + (size_t)(m0 + 16) * Dq;
    const bf16_t* B0  = Wt + (size_t)(n0 +  0) * Dq;
    const bf16_t* B1  = Wt + (size_t)(n0 + 16) * Dq;
    const bf16_t* B2  = Wt + (size_t)(n0 + 32) * Dq;
    const bf16_t* B3  = Wt + (size_t)(n0 + 48) * Dq;

    v16bf a0 = load_A_frag(Ar0, Dq, lane);
    v16bf a1 = load_A_frag(Ar1, Dq, lane);
    v16bf b0 = load_B_frag(B0, Dq, lane);
    v16bf b1 = load_B_frag(B1, Dq, lane);
    v16bf b2 = load_B_frag(B2, Dq, lane);
    v16bf b3 = load_B_frag(B3, Dq, lane);

    #pragma unroll 4
    for (int k0 = 0; k0 < Dq; k0 += 32) {
        const int kn = (k0 + 32 < Dq) ? (k0 + 32) : 0;   // last loads wrap (unused)
        __builtin_prefetch(Ar0 + (lane & 15) * Dq + k0 + 128, 0, 1);
        __builtin_prefetch(B0  + (lane & 15) * Dq + k0 + 128, 0, 1);
        v16bf a0n = load_A_frag(Ar0 + kn, Dq, lane);
        v16bf a1n = load_A_frag(Ar1 + kn, Dq, lane);
        v16bf b0n = load_B_frag(B0 + kn, Dq, lane);
        v16bf b1n = load_B_frag(B1 + kn, Dq, lane);
        v16bf b2n = load_B_frag(B2 + kn, Dq, lane);
        v16bf b3n = load_B_frag(B3 + kn, Dq, lane);

        acc[0] = wmma_bf16(a0, b0, acc[0]);
        acc[4] = wmma_bf16(a1, b0, acc[4]);
        acc[1] = wmma_bf16(a0, b1, acc[1]);
        acc[5] = wmma_bf16(a1, b1, acc[5]);
        acc[2] = wmma_bf16(a0, b2, acc[2]);
        acc[6] = wmma_bf16(a1, b2, acc[6]);
        acc[3] = wmma_bf16(a0, b3, acc[3]);
        acc[7] = wmma_bf16(a1, b3, acc[7]);

        a0 = a0n; a1 = a1n;
        b0 = b0n; b1 = b1n; b2 = b2n; b3 = b3n;
    }
}

// ---------------------------------------------------------------------------
// QKV GEMM: [8192,1024](bf16) x Wt[3072][1024](bf16, K-major) -> Q/K/Vt (bf16).
// Block = 8 waves: 2(M) x 4(N) -> block tile 64M x 256N.
// ---------------------------------------------------------------------------
__global__ __launch_bounds__(256) void qkv_gemm(const bf16_t* __restrict__ X,
                                                const bf16_t* __restrict__ Wt,
                                                bf16_t* __restrict__ Qb,
                                                bf16_t* __restrict__ Kb,
                                                bf16_t* __restrict__ Vtb) {
    const int lane = threadIdx.x & 31;
    const int wid  = threadIdx.x >> 5;
    const int m0 = blockIdx.x * 64 + (wid >> 2) * 32;
    const int n0 = blockIdx.y * 256 + (wid & 3) * 64;

    v8f acc[8] = {v8f{}, v8f{}, v8f{}, v8f{}, v8f{}, v8f{}, v8f{}, v8f{}};
    gemm_core_32x64(X, Wt, m0, n0, lane, acc);

    const int n_lane = lane & 15, hh = lane >> 4;
    const int b = m0 >> 11;                       // 64-row block never crosses b
    #pragma unroll
    for (int rb = 0; rb < 2; ++rb) {
        const int tbase = (m0 + rb * 16 + 8 * hh) & (Tq - 1);
        #pragma unroll
        for (int j = 0; j < 4; ++j) {
            v8f a = acc[rb * 4 + j];
            int n    = n0 + j * 16 + n_lane;
            int part = n >> 10;                   // 0:Q 1:K 2:V
            int nn   = n & (Dq - 1);
            int head = nn >> 6, dk = nn & 63;
            if (part == 2) {
                // Vt[bh][dk][t]: 8 consecutive t -> one 16B store
                v8bf pv;
                #pragma unroll
                for (int r = 0; r < 8; ++r) pv[r] = (bf16_t)a[r];
                *(v8bf*)(&Vtb[(((size_t)(b * Hq + head)) * DKq + dk) * Tq + tbase]) = pv;
            } else {
                bf16_t* dst = (part == 0) ? Qb : Kb;
                #pragma unroll
                for (int r = 0; r < 8; ++r)
                    dst[(((size_t)(b * Hq + head)) * Tq + tbase + r) * DKq + dk] = (bf16_t)a[r];
            }
        }
    }
}

// ---------------------------------------------------------------------------
// Cooperative K/V tile staging into LDS (one 32-key step), 256 threads.
//  K tile : 32 rows x 64 dk  -> kbuf rows padded to KT_STRIDE
//  Vt tile: 64 rows x 32 key -> vbuf rows padded to VT_STRIDE
// Async path uses GLOBAL_LOAD_ASYNC_TO_LDS_B128 (ASYNCcnt); fallback is
// global->reg->ds_store_b128 with identical layout.
// ---------------------------------------------------------------------------
__device__ __forceinline__ void stage_kv_tile(const bf16_t* __restrict__ Kh,
                                              const bf16_t* __restrict__ Vh,
                                              int kt, bf16_t* kbuf, bf16_t* vbuf,
                                              int tid) {
#if defined(ASYNC_COPY_OK)
    {
        const int row = tid >> 3, chunk = tid & 7;           // 32 x 8 x (8 bf16)
        const bf16_t* g = Kh + (size_t)(kt + row) * DKq + chunk * 8;
        bf16_t* l = kbuf + row * KT_STRIDE + chunk * 8;
        __builtin_amdgcn_global_load_async_to_lds_b128(
            (GAS v4i*)g, (LAS v4i*)l, 0, 0);
    }
    {
        const int row = tid >> 2, chunk = tid & 3;           // 64 x 4 x (8 bf16)
        const bf16_t* g = Vh + (size_t)row * Tq + kt + chunk * 8;
        bf16_t* l = vbuf + row * VT_STRIDE + chunk * 8;
        __builtin_amdgcn_global_load_async_to_lds_b128(
            (GAS v4i*)g, (LAS v4i*)l, 0, 0);
    }
#else
    {
        const int row = tid >> 3, chunk = tid & 7;
        v8bf d = *(const v8bf*)(Kh + (size_t)(kt + row) * DKq + chunk * 8);
        *(v8bf*)(kbuf + row * KT_STRIDE + chunk * 8) = d;
    }
    {
        const int row = tid >> 2, chunk = tid & 3;
        v8bf d = *(const v8bf*)(Vh + (size_t)row * Tq + kt + chunk * 8);
        *(v8bf*)(vbuf + row * VT_STRIDE + chunk * 8) = d;
    }
#endif
}

__device__ __forceinline__ void stage_wait() {
#if defined(ASYNC_COPY_OK)
    __builtin_amdgcn_s_wait_asynccnt(0);
#endif
    __syncthreads();
}

// ---------------------------------------------------------------------------
// Flash attention, block-cooperative: 8 waves share one (b,head); K/V tiles
// double-buffered in LDS; each wave owns one 16-query tile.
// ---------------------------------------------------------------------------
__global__ __launch_bounds__(256) void attn_fwd(const bf16_t* __restrict__ Qb,
                                                const bf16_t* __restrict__ Kb,
                                                const bf16_t* __restrict__ Vtb,
                                                bf16_t* __restrict__ Ob) {
    __shared__ __align__(16) bf16_t Kt[2][32 * KT_STRIDE];
    __shared__ __align__(16) bf16_t Vs[2][64 * VT_STRIDE];
    __shared__ __align__(16) bf16_t Pbuf[8][16 * VT_STRIDE]; // per-wave P tile

    const int tid  = threadIdx.x;
    const int lane = tid & 31;
    const int wid  = tid >> 5;
    const int bh   = blockIdx.y;           // b*H + head
    const int b    = bh >> 4, head = bh & 15;
    const int q0   = blockIdx.x * 128 + wid * 16;
    const int kmax = blockIdx.x * 128 + 128;   // exclusive key bound for block

    const bf16_t* Qh = Qb  + (size_t)bh * Tq * DKq;
    const bf16_t* Kh = Kb  + (size_t)bh * Tq * DKq;
    const bf16_t* Vh = Vtb + (size_t)bh * DKq * Tq;
    bf16_t* Pw = &Pbuf[wid][0];

    const int n_lane = lane & 15, hh = lane >> 4;
    const float scale = 0.125f;            // 1/sqrt(64)

    // Q fragments for this tile (K-dim 0..31 and 32..63), reused all loop.
    v16bf qf0 = load_A_frag(Qh + (size_t)q0 * DKq + 0,  DKq, lane);
    v16bf qf1 = load_A_frag(Qh + (size_t)q0 * DKq + 32, DKq, lane);

    v8f o[4] = {v8f{}, v8f{}, v8f{}, v8f{}};
    float mrow[8], lrow[8];
    #pragma unroll
    for (int r = 0; r < 8; ++r) { mrow[r] = -INFINITY; lrow[r] = 0.f; }

    // prologue: stage tile 0
    stage_kv_tile(Kh, Vh, 0, &Kt[0][0], &Vs[0][0], tid);
    stage_wait();

    int cur = 0;
    for (int kt = 0; kt < kmax; kt += 32, cur ^= 1) {
        // stage next tile while computing on current one
        if (kt + 32 < kmax)
            stage_kv_tile(Kh, Vh, kt + 32, &Kt[cur ^ 1][0], &Vs[cur ^ 1][0], tid);

        if (kt < q0 + 16) {                 // wave-uniform: EXEC stays all-1s
            const bf16_t* kb = &Kt[cur][0];
            const bf16_t* vb = &Vs[cur][0];

            // --- scores: S[16 x 32] as two 16x16 accumulators ---
            v16bf kf00 = load_B_frag(kb + 0,                  KT_STRIDE, lane);
            v16bf kf01 = load_B_frag(kb + 32,                 KT_STRIDE, lane);
            v16bf kf10 = load_B_frag(kb + 16 * KT_STRIDE,      KT_STRIDE, lane);
            v16bf kf11 = load_B_frag(kb + 16 * KT_STRIDE + 32, KT_STRIDE, lane);
            v8f s0 = v8f{}, s1 = v8f{};
            s0 = wmma_bf16(qf0, kf00, s0);
            s0 = wmma_bf16(qf1, kf01, s0);
            s1 = wmma_bf16(qf0, kf10, s1);
            s1 = wmma_bf16(qf1, kf11, s1);

            // --- online softmax, per owned row (M = r + 8*hh) ---
            #pragma unroll
            for (int r = 0; r < 8; ++r) {
                const int qrow = q0 + r + 8 * hh;
                float v0 = s0[r] * scale;
                float v1 = s1[r] * scale;
                if (kt + n_lane      > qrow) v0 = -INFINITY;   // causal mask
                if (kt + 16 + n_lane > qrow) v1 = -INFINITY;

                float tm = fmaxf(v0, v1);
                #pragma unroll
                for (int off = 8; off >= 1; off >>= 1)
                    tm = fmaxf(tm, __shfl_xor(tm, off, 32));

                float mnew  = fmaxf(mrow[r], tm);
                float alpha = __expf(mrow[r] - mnew);          // exp(-inf)=0 first pass
                float p0 = __expf(v0 - mnew);
                float p1 = __expf(v1 - mnew);

                float rs = p0 + p1;
                #pragma unroll
                for (int off = 8; off >= 1; off >>= 1)
                    rs += __shfl_xor(rs, off, 32);

                lrow[r] = lrow[r] * alpha + rs;
                mrow[r] = mnew;
                #pragma unroll
                for (int j = 0; j < 4; ++j) o[j][r] *= alpha;

                const int row = r + 8 * hh;
                Pw[row * VT_STRIDE + n_lane]      = (bf16_t)p0;
                Pw[row * VT_STRIDE + 16 + n_lane] = (bf16_t)p1;
            }

            // --- O += P[16x32] * V[32x64] from LDS ---
            v16bf pa = load_A_frag(Pw, VT_STRIDE, lane);
            o[0] = wmma_bf16(pa, load_B_frag(vb +  0 * VT_STRIDE, VT_STRIDE, lane), o[0]);
            o[1] = wmma_bf16(pa, load_B_frag(vb + 16 * VT_STRIDE, VT_STRIDE, lane), o[1]);
            o[2] = wmma_bf16(pa, load_B_frag(vb + 32 * VT_STRIDE, VT_STRIDE, lane), o[2]);
            o[3] = wmma_bf16(pa, load_B_frag(vb + 48 * VT_STRIDE, VT_STRIDE, lane), o[3]);
        }

        stage_wait();   // next tile landed (own async ops) + all waves done reading
    }

    // --- epilogue: normalize, store into [B,T,D] bf16 ---
    #pragma unroll
    for (int r = 0; r < 8; ++r) {
        float inv = 1.0f / lrow[r];
        int t = q0 + r + 8 * hh;
        #pragma unroll
        for (int j = 0; j < 4; ++j) {
            Ob[((size_t)(b * Tq + t)) * Dq + head * DKq + j * 16 + n_lane] =
                (bf16_t)(o[j][r] * inv);
        }
    }
}

// ---------------------------------------------------------------------------
// Projection GEMM: O[8192,1024](bf16) x WprojT[1024][1024](bf16) -> f32 out.
// ---------------------------------------------------------------------------
__global__ __launch_bounds__(256) void proj_gemm(const bf16_t* __restrict__ O,
                                                 const bf16_t* __restrict__ Wt,
                                                 float* __restrict__ Y) {
    const int lane = threadIdx.x & 31;
    const int wid  = threadIdx.x >> 5;
    const int m0 = blockIdx.x * 64 + (wid >> 2) * 32;
    const int n0 = blockIdx.y * 256 + (wid & 3) * 64;

    v8f acc[8] = {v8f{}, v8f{}, v8f{}, v8f{}, v8f{}, v8f{}, v8f{}, v8f{}};
    gemm_core_32x64(O, Wt, m0, n0, lane, acc);

    const int n_lane = lane & 15, hh = lane >> 4;
    #pragma unroll
    for (int rb = 0; rb < 2; ++rb) {
        #pragma unroll
        for (int j = 0; j < 4; ++j) {
            int n = n0 + j * 16 + n_lane;
            #pragma unroll
            for (int r = 0; r < 8; ++r) {
                int M = m0 + rb * 16 + r + 8 * hh;
                Y[(size_t)M * Dq + n] = acc[rb * 4 + j][r];
            }
        }
    }
}

// ---------------------------------------------------------------------------
extern "C" void kernel_launch(void* const* d_in, const int* in_sizes, int n_in,
                              void* d_out, int out_size, void* d_ws, size_t ws_size,
                              hipStream_t stream) {
    const float* x  = (const float*)d_in[0];   // [B,T,D]
    const float* Wa = (const float*)d_in[1];   // [D,3D]
    const float* Wp = (const float*)d_in[2];   // [D,D]
    float* y = (float*)d_out;                  // [B,T,D]

    char* ws = (char*)d_ws;
    const size_t MB = 1u << 20;
    bf16_t* Xbf = (bf16_t*)(ws);                 // 16 MB  [B*T, D]
    bf16_t* WaT = (bf16_t*)(ws + 16 * MB);       //  6 MB  [3D, D]
    bf16_t* WpT = (bf16_t*)(ws + 22 * MB);       //  2 MB  [D, D]
    bf16_t* Qb  = (bf16_t*)(ws + 24 * MB);       // 16 MB  [B,H,T,DK]
    bf16_t* Kb  = (bf16_t*)(ws + 40 * MB);       // 16 MB  [B,H,T,DK]
    bf16_t* Vt  = (bf16_t*)(ws + 56 * MB);       // 16 MB  [B,H,DK,T]
    bf16_t* Ob  = (bf16_t*)(ws + 72 * MB);       // 16 MB  [B*T, D]

    const int nx = Bq * Tq * Dq;
    cvt_f32_bf16<<<nx / 4 / 256, 256, 0, stream>>>(x, Xbf, nx);

    dim3 tb(32, 8);
    transpose_to_bf16<<<dim3(3 * Dq / 32, Dq / 32), tb, 0, stream>>>(Wa, WaT, Dq, 3 * Dq);
    transpose_to_bf16<<<dim3(Dq / 32, Dq / 32),     tb, 0, stream>>>(Wp, WpT, Dq, Dq);

    qkv_gemm<<<dim3((Bq * Tq) / 64, (3 * Dq) / 256), 256, 0, stream>>>(Xbf, WaT, Qb, Kb, Vt);

    attn_fwd<<<dim3(Tq / 128, Bq * Hq), 256, 0, stream>>>(Qb, Kb, Vt, Ob);

    proj_gemm<<<dim3((Bq * Tq) / 64, Dq / 256), 256, 0, stream>>>(Ob, WpT, y);
}
